// HierarchicalMemoryStack_21191368638923
// MI455X (gfx1250) — matrix operand order repaired and checked by reading, not compiled, not run on hardware
//
#include <hip/hip_runtime.h>
#include <math.h>

// HierarchicalMemoryStack on MI455X (gfx1250)
// L=4, B=128, IN=CTX=ST=512, R=C=256, D=1536

typedef __bf16 bf16_t;
typedef __attribute__((ext_vector_type(16))) __bf16 v16bf;
typedef __attribute__((ext_vector_type(8)))  float  v8f;
typedef __attribute__((ext_vector_type(4)))  float  f4;

#define LNUM  4
#define BNUM  128
#define DNUM  1536
#define STNUM 512
#define RNUM  256
#define CNUM  256

__device__ __forceinline__ v8f wmma_bf16(v16bf a, v16bf b, v8f c) {
  // D = A(16x32 bf16) * B(32x16 bf16) + C(16x16 f32)
  return __builtin_amdgcn_wmma_f32_16x16x32_bf16(false, a, false, b, (short)0, c,
                                                 false, false);
}

// split fp32 -> hi/lo bf16 pair (hi + lo ~= fp32; 3-product scheme recovers ~fp32)
__device__ __forceinline__ void cvt16(const float* buf, v16bf& hi, v16bf& lo) {
#pragma unroll
  for (int i = 0; i < 16; ++i) {
    float f = buf[i];
    bf16_t h = (bf16_t)f;
    hi[i] = h;
    lo[i] = (bf16_t)(f - (float)h);
  }
}

// A fragment (contiguous k order, same as frag_b): lane holds row M=lane&15;
// elems 0..7  = K = kb + half*16 + {0..7}
// elems 8..15 = K = kb + half*16 + 8 + {0..7}      (half = lane>>4)
__device__ __forceinline__ void frag_a(const float* rowp, int half, v16bf& hi, v16bf& lo) {
  float buf[16];
  const float* p0 = rowp + half * 16;
  *(f4*)(buf + 0)  = *(const f4*)(p0);
  *(f4*)(buf + 4)  = *(const f4*)(p0 + 4);
  const float* p1 = rowp + half * 16 + 8;
  *(f4*)(buf + 8)  = *(const f4*)(p1);
  *(f4*)(buf + 12) = *(const f4*)(p1 + 4);
  cvt16(buf, hi, lo);
}

// B fragment (B 32x16): lane holds col N=lane&15; elem i = B[half*16 + i, N]
// B = W^T so this is 16 CONSECUTIVE floats of W's row (o_base+N) at kb+half*16.
__device__ __forceinline__ void frag_b(const float* p, v16bf& hi, v16bf& lo) {
  float buf[16];
#pragma unroll
  for (int i = 0; i < 4; ++i)
    *(f4*)(buf + 4 * i) = *(const f4*)(p + 4 * i);
  cvt16(buf, hi, lo);
}

__device__ __forceinline__ float sigm(float x) { return 1.0f / (1.0f + __expf(-x)); }

// ---------------------------------------------------------------------------
// Kernel 1: gates GEMM + fused gating -> h_new  (written directly to d_out)
// grid (ST/16, L), block 256 (8 waves). wave w -> batch row-tile w, block.x -> col tile.
// ---------------------------------------------------------------------------
__global__ __launch_bounds__(256) void hms_gates(
    const float* __restrict__ x, const float* __restrict__ cc, const float* __restrict__ h,
    const float* __restrict__ phi_w, const float* __restrict__ phi_b,
    const float* __restrict__ alpha_w, const float* __restrict__ alpha_b,
    const float* __restrict__ beta_w, const float* __restrict__ beta_b,
    float* __restrict__ h_new) {
  const int l    = blockIdx.y;
  const int colt = blockIdx.x;
  const int wave = threadIdx.x >> 5;
  const int lane = threadIdx.x & 31;
  const int n    = lane & 15;
  const int half = lane >> 4;
  const int row  = wave * 16 + n;     // A-matrix row (batch)
  const int ocol = colt * 16 + n;     // B-matrix col (output feature)

  const float* wp = phi_w   + (size_t)(l * STNUM + ocol) * DNUM + half * 16;
  const float* wa = alpha_w + (size_t)(l * STNUM + ocol) * DNUM + half * 16;
  const float* wb = beta_w  + (size_t)(l * STNUM + ocol) * DNUM + half * 16;

  const float* arow_x = x  + (size_t)row * 512;
  const float* arow_c = cc + (size_t)row * 512;
  const float* arow_h = h  + (size_t)(l * BNUM + row) * 512;

  v8f accp = {}, acca = {}, accb = {};

  for (int kb = 0; kb < DNUM; kb += 32) {
    const int seg = kb >> 9;          // gi = [x | c | h[l]], 512 each (wave-uniform)
    const int kk  = kb & 511;
    const float* arow = (seg == 0) ? (arow_x + kk)
                      : (seg == 1) ? (arow_c + kk)
                                   : (arow_h + kk);
    v16bf ahi, alo;
    frag_a(arow, half, ahi, alo);

    v16bf bhi, blo;
    frag_b(wp + kb, bhi, blo);
    accp = wmma_bf16(ahi, bhi, accp);
    accp = wmma_bf16(alo, bhi, accp);
    accp = wmma_bf16(ahi, blo, accp);

    frag_b(wa + kb, bhi, blo);
    acca = wmma_bf16(ahi, bhi, acca);
    acca = wmma_bf16(alo, bhi, acca);
    acca = wmma_bf16(ahi, blo, acca);

    frag_b(wb + kb, bhi, blo);
    accb = wmma_bf16(ahi, bhi, accb);
    accb = wmma_bf16(alo, bhi, accb);
    accb = wmma_bf16(ahi, blo, accb);
  }

  // epilogue: h_new = sigmoid(alpha)*tanh(phi) + sigmoid(beta)*h
  const float bp = phi_b[l * STNUM + ocol];
  const float ba = alpha_b[l * STNUM + ocol];
  const float bb = beta_b[l * STNUM + ocol];
  const int rbase = wave * 16 + half * 8;   // C/D layout: M = vgpr_i + half*8
#pragma unroll
  for (int i = 0; i < 8; ++i) {
    const int r = rbase + i;
    const size_t idx = (size_t)(l * BNUM + r) * STNUM + ocol;
    const float ph = accp[i] + bp;
    const float al = sigm(acca[i] + ba);
    const float be = sigm(accb[i] + bb);
    h_new[idx] = al * tanhf(ph) + be * h[idx];
  }
}

// ---------------------------------------------------------------------------
// Kernel 2: u/v GEMM from wv = [h_new | x | c]; u|v packed over 32 col tiles.
// grid (32, L), block 256.
// ---------------------------------------------------------------------------
__global__ __launch_bounds__(256) void hms_uv(
    const float* __restrict__ h_new, const float* __restrict__ x, const float* __restrict__ cc,
    const float* __restrict__ u_w, const float* __restrict__ u_b,
    const float* __restrict__ v_w, const float* __restrict__ v_b,
    float* __restrict__ u_out, float* __restrict__ v_out) {
  const int l    = blockIdx.y;
  const int colt = blockIdx.x;            // <16 -> u, else v (block-uniform)
  const int wave = threadIdx.x >> 5;
  const int lane = threadIdx.x & 31;
  const int n    = lane & 15;
  const int half = lane >> 4;
  const int row  = wave * 16 + n;
  const bool is_u = colt < 16;
  const int olocal = (colt & 15) * 16 + n;

  const float* W = (is_u ? u_w : v_w) + (size_t)(l * RNUM + olocal) * DNUM + half * 16;

  const float* arow_h = h_new + (size_t)(l * BNUM + row) * 512;
  const float* arow_x = x  + (size_t)row * 512;
  const float* arow_c = cc + (size_t)row * 512;

  v8f acc = {};
  for (int kb = 0; kb < DNUM; kb += 32) {
    const int seg = kb >> 9;              // wv = [h_new | x | c]
    const int kk  = kb & 511;
    const float* arow = (seg == 0) ? (arow_h + kk)
                      : (seg == 1) ? (arow_x + kk)
                                   : (arow_c + kk);
    v16bf ahi, alo, bhi, blo;
    frag_a(arow, half, ahi, alo);
    frag_b(W + kb, bhi, blo);
    acc = wmma_bf16(ahi, bhi, acc);
    acc = wmma_bf16(alo, bhi, acc);
    acc = wmma_bf16(ahi, blo, acc);
  }

  const float bias = (is_u ? u_b : v_b)[l * RNUM + olocal];
  float* outp = is_u ? u_out : v_out;
  const int rbase = wave * 16 + half * 8;
#pragma unroll
  for (int i = 0; i < 8; ++i) {
    const int r = rbase + i;
    outp[(size_t)(l * BNUM + r) * RNUM + olocal] = acc[i] + bias;
  }
}

// ---------------------------------------------------------------------------
// Kernel 3: streaming EMA memory update + rank-1 write.
// M_new = (1-g)*M + 0.5g*(M_up + M_dn) + e*u (x) v
// grid 4096 blocks x 256 threads; per block: one (l,b) tile slice, coalesced f4.
// Non-temporal stores: M_new is never re-read, keep L2 for M.
// ---------------------------------------------------------------------------
__global__ __launch_bounds__(256) void hms_mupdate(
    const float* __restrict__ M, const float* __restrict__ u, const float* __restrict__ v,
    const float* __restrict__ gamma_logits, const float* __restrict__ eta_logits,
    float* __restrict__ M_new) {
  const int part = blockIdx.x & 7;
  const int tile = blockIdx.x >> 3;       // tile = l*128 + b, 0..511
  const int l = tile >> 7;
  const int b = tile & 127;
  const int lup = (l == 0) ? 0 : l - 1;
  const int ldn = (l == LNUM - 1) ? l : l + 1;

  const float g   = sigm(gamma_logits[l]);
  const float e   = sigm(eta_logits[l]);
  const float omg = 1.0f - g;
  const float hg  = 0.5f * g;

  const int tid = threadIdx.x;
  const int j4  = tid & 63;               // fixed column group per thread
  const f4 vv   = *(const f4*)(v + (size_t)tile * CNUM + j4 * 4);

  const size_t self_base = (size_t)tile * RNUM;
  const size_t up_base   = (size_t)(lup * BNUM + b) * RNUM;
  const size_t dn_base   = (size_t)(ldn * BNUM + b) * RNUM;

#pragma unroll
  for (int it = 0; it < 8; ++it) {
    const int iter = part * 8 + it;
    const int r = iter * 4 + (tid >> 6);  // fixed row per thread per iteration
    const float uv = u[(size_t)tile * RNUM + r] * e;
    const size_t ro = (size_t)r * CNUM + j4 * 4;
    const f4 m  = *(const f4*)(M + (self_base * CNUM + ro));
    const f4 mu = *(const f4*)(M + (up_base   * CNUM + ro));
    const f4 md = *(const f4*)(M + (dn_base   * CNUM + ro));
    const f4 out = omg * m + hg * (mu + md) + uv * vv;
    __builtin_nontemporal_store(out, (f4*)(M_new + (self_base * CNUM + ro)));
  }
}

// ---------------------------------------------------------------------------
extern "C" void kernel_launch(void* const* d_in, const int* in_sizes, int n_in,
                              void* d_out, int out_size, void* d_ws, size_t ws_size,
                              hipStream_t stream) {
  const float* x       = (const float*)d_in[0];
  const float* c       = (const float*)d_in[1];
  const float* h       = (const float*)d_in[2];
  const float* M       = (const float*)d_in[3];
  const float* phi_w   = (const float*)d_in[4];
  const float* phi_b   = (const float*)d_in[5];
  const float* alpha_w = (const float*)d_in[6];
  const float* alpha_b = (const float*)d_in[7];
  const float* beta_w  = (const float*)d_in[8];
  const float* beta_b  = (const float*)d_in[9];
  const float* u_w     = (const float*)d_in[10];
  const float* u_b     = (const float*)d_in[11];
  const float* v_w     = (const float*)d_in[12];
  const float* v_b     = (const float*)d_in[13];
  const float* gamma_l = (const float*)d_in[14];
  const float* eta_l   = (const float*)d_in[15];

  float* h_new = (float*)d_out;                                 // (L,B,ST)
  float* M_new = (float*)d_out + (size_t)LNUM * BNUM * STNUM;   // (L,B,R,C)
  float* u_out = (float*)d_ws;                                  // (L,B,R)
  float* v_out = u_out + (size_t)LNUM * BNUM * RNUM;            // (L,B,C)

  hms_gates<<<dim3(STNUM / 16, LNUM), 256, 0, stream>>>(
      x, c, h, phi_w, phi_b, alpha_w, alpha_b, beta_w, beta_b, h_new);
  hms_uv<<<dim3(32, LNUM), 256, 0, stream>>>(
      h_new, x, c, u_w, u_b, v_w, v_b, u_out, v_out);
  hms_mupdate<<<dim3(4096), 256, 0, stream>>>(
      M, u_out, v_out, gamma_l, eta_l, M_new);
}
